// S_DarkNet_Tiny_v2_67405216744269
// MI455X (gfx1250) — matrix operand order repaired and checked by reading, not compiled, not run on hardware
//
#include <hip/hip_runtime.h>
#include <hip/hip_bf16.h>
#include <stdint.h>

typedef __attribute__((ext_vector_type(16))) _Float16 v16h;
typedef __attribute__((ext_vector_type(8)))  _Float16 v8h;
typedef __attribute__((ext_vector_type(8)))  float    v8f;
typedef unsigned int u32x4 __attribute__((ext_vector_type(4)));
typedef int          i32x4 __attribute__((ext_vector_type(4)));
typedef int          i32x8 __attribute__((ext_vector_type(8)));

#define TT 7   // timesteps

// ---------------- BN/bias folding: scale = g/sqrt(v+eps), bias = b*scale + beta - mean*scale
__global__ void bn_coeffs_k(const float* __restrict__ g, const float* __restrict__ be,
                            const float* __restrict__ mn, const float* __restrict__ vr,
                            const float* __restrict__ bconv, float* __restrict__ scale,
                            float* __restrict__ bias, int C) {
    int c = blockIdx.x * blockDim.x + threadIdx.x;
    if (c >= C) return;
    float s = g[c] * rsqrtf(vr[c] + 1e-5f);
    scale[c] = s;
    bias[c]  = bconv[c] * s + be[c] - mn[c] * s;
}

// ---------------- pack fp32 weights [cout][K] -> f16 padded [coutPad][Kpad] (zero fill)
__global__ void pack_w_k(const float* __restrict__ w, _Float16* __restrict__ dst,
                         int cout, int K, int Kpad, int total) {
    int i = blockIdx.x * blockDim.x + threadIdx.x;
    if (i >= total) return;
    int r = i / Kpad, k = i - r * Kpad;
    float v = (r < cout && k < K) ? w[(size_t)r * K + k] : 0.f;
    dst[i] = (_Float16)v;
}

// ---------------- layer 1: direct fp32 conv 3->16 s2 p1, computed once (identical across T), BN fused
__global__ void conv1_direct_k(const float* __restrict__ x, const float* __restrict__ w,
                               const float* __restrict__ scale, const float* __restrict__ bias,
                               float* __restrict__ out) {
    const int HW = 208 * 208;
    int i = blockIdx.x * blockDim.x + threadIdx.x;
    if (i >= 4 * 16 * HW) return;
    int n   = i / (16 * HW);
    int rem = i - n * 16 * HW;
    int co  = rem / HW;
    int sp  = rem - co * HW;
    int oh = sp / 208, ow = sp - oh * 208;
    float acc = 0.f;
    #pragma unroll
    for (int ci = 0; ci < 3; ++ci)
        #pragma unroll
        for (int kh = 0; kh < 3; ++kh) {
            int ih = oh * 2 - 1 + kh;
            if ((unsigned)ih >= 416u) continue;
            #pragma unroll
            for (int kw = 0; kw < 3; ++kw) {
                int iw = ow * 2 - 1 + kw;
                if ((unsigned)iw >= 416u) continue;
                acc += x[((n * 3 + ci) * 416 + ih) * 416 + iw] *
                       w[((co * 3 + ci) * 3 + kh) * 3 + kw];
            }
        }
    out[i] = acc * scale[co] + bias[co];
}

// ---------------- integrate & fire with inhibition; emits (spike - inhibit)*thr as f16
__global__ void if_spike_k(const float* __restrict__ pre, _Float16* __restrict__ spk,
                           const float* __restrict__ thrp, int CHW, int bcast) {
    int i = blockIdx.x * blockDim.x + threadIdx.x;
    if (i >= 4 * CHW) return;
    int b = i / CHW, r = i - b * CHW;
    float thr = thrp[0];
    float mem = 0.5f * thr, ssum = 0.f;
    #pragma unroll
    for (int t = 0; t < TT; ++t) {
        float a = bcast ? pre[i] : pre[(size_t)(b * TT + t) * CHW + r];
        mem += a;
        float sp = (mem >= thr) ? 1.f : 0.f;
        mem -= thr * sp;  ssum += sp;
        float inh = ((mem <= -0.001f) && (ssum > 0.f)) ? 1.f : 0.f;
        mem += thr * inh; ssum -= inh;
        spk[(size_t)(b * TT + t) * CHW + r] = (_Float16)((sp - inh) * thr);
    }
}

// ---------------- implicit-GEMM conv: WMMA f16 + TDM-staged weight panels
// block = 128 threads (4 waves); block tile = (64*MW)(M) x 64(N)
// A panel (64*MW rows x 32 K, f16) is DMA'd into LDS by the Tensor Data Mover,
// double-buffered, with TDM LDS-padding giving an 80B row pitch (conflict-free ds reads).
template <int MW>
__launch_bounds__(128)
__global__ void conv_wmma_k(const _Float16* __restrict__ act, const _Float16* __restrict__ wpk,
                            const float* __restrict__ scale, const float* __restrict__ bias,
                            float* __restrict__ out,
                            int cin, int Hi, int Wi, int cout, int Ho, int Wo,
                            int ks, int stride, int pad, int K, int Kpad, int NP) {
    constexpr int RowsT   = 64 * MW;          // weight rows per block
    constexpr int ARow    = 40;               // f16 per A row in LDS (32 data + 4-DWORD TDM pad)
    constexpr int BtBytes = 64 * 40 * 2;      // B tile bytes
    __shared__ __align__(16) unsigned char smem[BtBytes + 2 * RowsT * ARow * 2];
    _Float16 (*Bt)[40] = reinterpret_cast<_Float16(*)[40]>(smem);
    _Float16* At       = reinterpret_cast<_Float16*>(smem + BtBytes);

    const int tid  = threadIdx.x;
    const int wv   = tid >> 5;
    const int lane = tid & 31;
    const int kk   = lane;                    // gather K slot
    const int pBase = blockIdx.x * 64;
    const int coBlk = blockIdx.y * RowsT;
    const int HiWi = Hi * Wi, HoWo = Ho * Wo;
    const int ksq = ks * ks;

    // issue one TDM 2D-tile load: rows [coBlk, coBlk+RowsT) x K [k0c, k0c+32) -> At[buf]
    auto issue_tdm = [&](int k0c, int buf) {
        uint64_t ga = (uint64_t)(uintptr_t)wpk +
                      ((uint64_t)(unsigned)coBlk * (unsigned)Kpad + (unsigned)k0c) * 2u;
        unsigned ldsOff = (unsigned)BtBytes + (unsigned)(buf * RowsT * ARow * 2);
        u32x4 g0 = { 1u,                                   // count=1 valid descriptor
                     ldsOff,                               // lds_addr
                     (unsigned)ga,                         // global_addr[31:0]
                     (unsigned)((ga >> 32) & 0x1FFFFFFu) | (2u << 30) }; // addr[56:32] | type=2
        i32x8 g1 = { (int)((1u << 16) | (1u << 20) | (3u << 22) | (3u << 25)),
                     //   data_size=2B  pad_en     pad_int=16DW  pad_amt=4DW
                     (int)(32u << 16),                     // tensor_dim0 = 32
                     (int)((unsigned)RowsT << 16),         // tensor_dim1 = RowsT
                     (int)(32u << 16),                     // tile_dim0 = 32
                     (int)RowsT,                           // tile_dim1 = RowsT (tile_dim2=0)
                     (int)Kpad,                            // tensor_dim0_stride = row pitch
                     0, 0 };
        i32x4 z4 = { 0, 0, 0, 0 };
#if defined(__clang_major__) && __clang_major__ >= 23
        i32x8 z8 = { 0, 0, 0, 0, 0, 0, 0, 0 };
        __builtin_amdgcn_tensor_load_to_lds(g0, g1, z4, z4, z8, 0);
#else
        __builtin_amdgcn_tensor_load_to_lds(g0, g1, z4, z4, 0);
#endif
    };

    // per-thread im2col position info (16 N slots: n = wv + 4*i)
    int gimg[16], gih[16], giw[16]; bool gv[16];
    #pragma unroll
    for (int i = 0; i < 16; ++i) {
        int p = pBase + wv + 4 * i;
        gv[i] = (p < NP);
        int img = 0, ihb = 0, iwb = 0;
        if (gv[i]) {
            img = p / HoWo; int sp = p - img * HoWo;
            int oh = sp / Wo, ow = sp - oh * Wo;
            ihb = oh * stride - pad; iwb = ow * stride - pad;
        }
        gimg[i] = img; gih[i] = ihb; giw[i] = iwb;
    }

    const int hi8 = (lane & 16) ? 8 : 0;

    v8f acc[MW][4];
    #pragma unroll
    for (int m = 0; m < MW; ++m)
        #pragma unroll
        for (int s = 0; s < 4; ++s)
            #pragma unroll
            for (int j = 0; j < 8; ++j) acc[m][s][j] = 0.f;

    if (wv == 0) issue_tdm(0, 0);             // prologue: stage first A panel

    for (int k0 = 0; k0 < Kpad; k0 += 32) {
        const int  buf     = (k0 >> 5) & 1;
        const bool hasNext = (k0 + 32 < Kpad);
        int k = k0 + kk;
        int ci, kh, kw;
        if (ksq == 9) { ci = k / 9; int rm = k - ci * 9; kh = rm / 3; kw = rm - kh * 3; }
        else          { ci = k; kh = 0; kw = 0; }
        bool kval = (k < K);

        __syncthreads();                       // prior iter's LDS reads are done
        if (wv == 0 && hasNext) issue_tdm(k0 + 32, buf ^ 1);

        #pragma unroll
        for (int i = 0; i < 16; ++i) {
            _Float16 v = (_Float16)0.f;
            if (kval && gv[i]) {
                int ih = gih[i] + kh, iw = giw[i] + kw;
                if ((unsigned)ih < (unsigned)Hi && (unsigned)iw < (unsigned)Wi)
                    v = act[(size_t)(gimg[i] * cin + ci) * HiWi + ih * Wi + iw];
            }
            Bt[wv + 4 * i][kk] = v;
        }
        if (hasNext) __builtin_amdgcn_s_wait_tensorcnt(1);   // current panel landed
        else         __builtin_amdgcn_s_wait_tensorcnt(0);   // last panel landed
        __syncthreads();

        // A fragments from LDS panel: row-major, 80B pitch; K halves split at lane 16
        v16h a[MW];
        #pragma unroll
        for (int m = 0; m < MW; ++m) {
            const _Float16* arow = At + buf * RowsT * ARow +
                                   (m * 64 + wv * 16 + (lane & 15)) * ARow;
            v8h alo = *(const v8h*)(arow + hi8);
            v8h ahi = *(const v8h*)(arow + hi8 + 16);
            #pragma unroll
            for (int j = 0; j < 8; ++j) { a[m][j] = alo[j]; a[m][j + 8] = ahi[j]; }
        }

        const int kb = (lane < 16) ? 0 : 8;    // B: lanes 0-15 K=0..7,16..23, lanes 16-31 K=8..15,24..31
        #pragma unroll
        for (int s = 0; s < 4; ++s) {
            const _Float16* bp = &Bt[s * 16 + (lane & 15)][kb];
            v8h blo = *(const v8h*)bp;
            v8h bhi = *(const v8h*)(bp + 16);
            v16h b;
            #pragma unroll
            for (int j = 0; j < 8; ++j) { b[j] = blo[j]; b[j + 8] = bhi[j]; }
            #pragma unroll
            for (int m = 0; m < MW; ++m)
                acc[m][s] = __builtin_amdgcn_wmma_f32_16x16x32_f16(
                    false, a[m], false, b, (short)0, acc[m][s], false, false);
        }
    }

    // C layout: VGPR r -> M = r + 8*(lane>=16); N = lane%16. Fused affine epilogue.
    #pragma unroll
    for (int s = 0; s < 4; ++s) {
        int p = pBase + s * 16 + (lane & 15);
        if (p >= NP) continue;
        int img = p / HoWo; int sp = p - img * HoWo;
        #pragma unroll
        for (int m = 0; m < MW; ++m)
            #pragma unroll
            for (int r = 0; r < 8; ++r) {
                int co = coBlk + m * 64 + wv * 16 + r + ((lane & 16) ? 8 : 0);
                if (co < cout) {
                    float sc = scale ? scale[co] : 1.f;
                    float bi = bias  ? bias[co]  : 0.f;
                    out[(size_t)(img * cout + co) * HoWo + sp] = acc[m][s][r] * sc + bi;
                }
            }
    }
}

// ---------------- rate decode: out[b,*] = mean over t of pred[(b*7+t),*]
__global__ void pred_reduce_k(const float* __restrict__ pb, float* __restrict__ out) {
    const int PS = 125 * 169;
    int i = blockIdx.x * blockDim.x + threadIdx.x;
    if (i >= 4 * PS) return;
    int b = i / PS, r = i - b * PS;
    float s = 0.f;
    #pragma unroll
    for (int t = 0; t < TT; ++t) s += pb[(size_t)(b * TT + t) * PS + r];
    out[i] = s * (1.f / (float)TT);
}

// =====================================================================
extern "C" void kernel_launch(void* const* d_in, const int* in_sizes, int n_in,
                              void* d_out, int out_size, void* d_ws, size_t ws_size,
                              hipStream_t stream) {
    struct LayerCfg { int cin, cout, stride, Hi, Wi, Ho, Wo; };
    static const LayerCfg L[8] = {
        {3,   16,  2, 416, 416, 208, 208}, {16,  32,  2, 208, 208, 104, 104},
        {32,  64,  2, 104, 104, 52,  52 }, {64,  128, 2, 52,  52,  26,  26 },
        {128, 256, 2, 26,  26,  13,  13 }, {256, 512, 1, 13,  13,  13,  13 },
        {512, 1024,1, 13,  13,  13,  13 }, {1024,1024,1, 13,  13,  13,  13 }};
    static const int MWL[9] = {2 /*pred*/, 0, 1, 1, 2, 2, 2, 2, 2};

    auto Wp  = [&](int li) { return (const float*)d_in[1 + (li - 1) * 7 + 0]; };
    auto Bp  = [&](int li) { return (const float*)d_in[1 + (li - 1) * 7 + 1]; };
    auto Gp  = [&](int li) { return (const float*)d_in[1 + (li - 1) * 7 + 2]; };
    auto Bep = [&](int li) { return (const float*)d_in[1 + (li - 1) * 7 + 3]; };
    auto Mp  = [&](int li) { return (const float*)d_in[1 + (li - 1) * 7 + 4]; };
    auto Vp  = [&](int li) { return (const float*)d_in[1 + (li - 1) * 7 + 5]; };
    auto Thp = [&](int li) { return (const float*)d_in[1 + (li - 1) * 7 + 6]; };
    const float* x      = (const float*)d_in[0];
    const float* w_pred = (const float*)d_in[57];
    const float* b_pred = (const float*)d_in[58];

    char* ws = (char*)d_ws;
    size_t off = 0;
    auto carve = [&](size_t bytes) -> char* {
        off = (off + 255) & ~(size_t)255;
        char* p = ws + off; off += bytes; return p;
    };
    float*    scb    = (float*)carve(8 * 2048 * sizeof(float));
    float*    preact = (float*)carve((size_t)9691136 * sizeof(float));        // 28*32*104*104
    _Float16* sA     = (_Float16*)carve((size_t)19382272 * sizeof(_Float16)); // 28*16*208*208
    _Float16* sB     = (_Float16*)carve((size_t)9691136  * sizeof(_Float16)); // 28*32*104*104

    _Float16* wpk[9];
    int       KpadA[9], rowsA[9];
    for (int li = 2; li <= 8; ++li) {
        int K = L[li - 1].cin * 9;
        int Kpad = ((K + 31) / 32) * 32;
        int mrow = 64 * MWL[li];
        int rows = ((L[li - 1].cout + mrow - 1) / mrow) * mrow;
        KpadA[li] = Kpad; rowsA[li] = rows;
        wpk[li] = (_Float16*)carve((size_t)rows * Kpad * sizeof(_Float16));
    }
    { KpadA[0] = 1024; rowsA[0] = 128;
      wpk[0] = (_Float16*)carve((size_t)128 * 1024 * sizeof(_Float16)); }

    for (int li = 1; li <= 8; ++li) {
        int C = L[li - 1].cout;
        bn_coeffs_k<<<dim3((C + 63) / 64), dim3(64), 0, stream>>>(
            Gp(li), Bep(li), Mp(li), Vp(li), Bp(li),
            scb + (li - 1) * 2048, scb + (li - 1) * 2048 + 1024, C);
    }

    for (int li = 2; li <= 8; ++li) {
        int K = L[li - 1].cin * 9;
        int total = rowsA[li] * KpadA[li];
        pack_w_k<<<dim3((total + 255) / 256), dim3(256), 0, stream>>>(
            Wp(li), wpk[li], L[li - 1].cout, K, KpadA[li], total);
    }
    { int total = 128 * 1024;
      pack_w_k<<<dim3((total + 255) / 256), dim3(256), 0, stream>>>(
          w_pred, wpk[0], 125, 1024, 1024, total); }

    { int n = 4 * 16 * 208 * 208;
      conv1_direct_k<<<dim3((n + 255) / 256), dim3(256), 0, stream>>>(
          x, Wp(1), scb + 0, scb + 1024, preact);
      int CHW = 16 * 208 * 208;
      if_spike_k<<<dim3((4 * CHW + 255) / 256), dim3(256), 0, stream>>>(
          preact, sA, Thp(1), CHW, 1); }

    for (int li = 2; li <= 8; ++li) {
        const LayerCfg& c = L[li - 1];
        const _Float16* in  = (li % 2 == 0) ? sA : sB;
        _Float16*       osp = (li % 2 == 0) ? sB : sA;
        int K = c.cin * 9, Kpad = KpadA[li];
        int NP = 28 * c.Ho * c.Wo;
        dim3 grid((NP + 63) / 64, rowsA[li] / (64 * MWL[li]));
        if (MWL[li] == 1)
            conv_wmma_k<1><<<grid, dim3(128), 0, stream>>>(
                in, wpk[li], scb + (li - 1) * 2048, scb + (li - 1) * 2048 + 1024, preact,
                c.cin, c.Hi, c.Wi, c.cout, c.Ho, c.Wo, 3, c.stride, 1, K, Kpad, NP);
        else
            conv_wmma_k<2><<<grid, dim3(128), 0, stream>>>(
                in, wpk[li], scb + (li - 1) * 2048, scb + (li - 1) * 2048 + 1024, preact,
                c.cin, c.Hi, c.Wi, c.cout, c.Ho, c.Wo, 3, c.stride, 1, K, Kpad, NP);
        int CHW = c.cout * c.Ho * c.Wo;
        if_spike_k<<<dim3((4 * CHW + 255) / 256), dim3(256), 0, stream>>>(
            preact, osp, Thp(li), CHW, 0);
    }

    { int NP = 28 * 169;
      dim3 grid((NP + 63) / 64, 1);
      conv_wmma_k<2><<<grid, dim3(128), 0, stream>>>(
          sB, wpk[0], nullptr, b_pred, preact,
          1024, 13, 13, 125, 13, 13, 1, 1, 0, 1024, 1024, NP);
      int n = 4 * 125 * 169;
      pred_reduce_k<<<dim3((n + 255) / 256), dim3(256), 0, stream>>>(
          preact, (float*)d_out); }
}
